// MultiheadAttention_37383395345076
// MI455X (gfx1250) — compile-verified, matrix operand order checked by reading, not run on hardware
//
#include <hip/hip_runtime.h>

#define DEV __device__ __forceinline__

typedef __attribute__((ext_vector_type(8)))  __bf16 v8bf;
typedef __attribute__((ext_vector_type(16))) __bf16 v16bf;
typedef __attribute__((ext_vector_type(8)))  float  v8f;
typedef __attribute__((ext_vector_type(4)))  float  v4f;
typedef __attribute__((ext_vector_type(4)))  unsigned int v4u;
typedef __attribute__((ext_vector_type(8)))  unsigned int v8u;

// Problem constants (fixed by the reference).
constexpr int Bb = 4, Nn = 2048, Dd = 1024, Hh = 16, DH = 64;
constexpr int Tt = Bb * Nn;              // 8192 tokens

DEV v16bf cat16(v8bf lo, v8bf hi) {
  return __builtin_shufflevector(lo, hi, 0,1,2,3,4,5,6,7,8,9,10,11,12,13,14,15);
}
DEV v8bf ld8bf(const __bf16* p) { return *(const v8bf*)p; }

// 8 consecutive f32 -> v8bf (in-register down-convert; avoids a conversion pass)
DEV v8bf cvt8(const float* p) {
  v4f a = *(const v4f*)p;
  v4f b = *(const v4f*)(p + 4);
  v8bf r;
#pragma unroll
  for (int i = 0; i < 4; ++i) { r[i] = (__bf16)a[i]; r[i + 4] = (__bf16)b[i]; }
  return r;
}

DEV v8f wmma_bf16(v16bf a, v16bf b, v8f c) {
  // D = A(16x32) x B(32x16) + C, f32 accumulate
  return __builtin_amdgcn_wmma_f32_16x16x32_bf16(false, a, false, b, (short)0, c,
                                                 false, false);
}

// Raw hardware exp2 (v_exp_f32, no denorm-input fixup): flash-attention
// arguments are <= 0 and well within native range.
DEV float fast_exp2(float x) { return __builtin_amdgcn_exp2f(x); }

// ---------------------------------------------------------------------------
// Tensor Data Mover: issue a 2-D tile DMA (global -> LDS).
// D# per CDNA5 ISA: group0 = {count=1, lds_addr, global_addr,
// type=2}; group1 = {data_size=8B, tensor/tile dims, dim0 stride}.
// Units are 8-byte elements (data_size code 3). 1-D copies use tile_d1=1.
// ---------------------------------------------------------------------------
DEV void tdm_load_to_lds(unsigned int lds_off, const void* gptr,
                         unsigned int tensor_d0, unsigned int tensor_d1,
                         unsigned int tile_d0, unsigned int tile_d1,
                         unsigned long long stride0_units) {
  unsigned long long ga = (unsigned long long)(size_t)gptr;
  v4u g0;
  g0[0] = 1u;                                            // count=1, user mode
  g0[1] = lds_off;                                       // lds_addr [63:32]
  g0[2] = (unsigned int)ga;                              // global_addr lo
  g0[3] = (unsigned int)((ga >> 32) & 0x1FFFFFFu) | (2u << 30);  // hi | type=2
  v8u g1;
  g1[0] = (3u << 16);                                    // data_size = 8 bytes
  g1[1] = (tensor_d0 & 0xFFFFu) << 16;                   // tensor_dim0 lo16
  g1[2] = (tensor_d0 >> 16) | ((tensor_d1 & 0xFFFFu) << 16);
  g1[3] = (tensor_d1 >> 16) | (tile_d0 << 16);           // tile_dim0
  g1[4] = tile_d1 & 0xFFFFu;                             // tile_dim1 (dim2=0)
  g1[5] = (unsigned int)stride0_units;                   // tensor_dim0_stride
  g1[6] = (unsigned int)(stride0_units >> 32) & 0xFFFFu; // (dim1_stride=0)
  g1[7] = 0u;
  asm volatile("tensor_load_to_lds %0, %1" : : "s"(g0), "s"(g1) : "memory");
}

// ---------------------------------------------------------------------------
// Projection GEMM, computed transposed: yT(j, t) = sum_k W[j][k] * X[t][k].
// A-tile = 16 W rows, B-tile = 16 X rows; both use the canonical 16-bit
// operand mapping: element e -> k = (e<8?0:16) + half*8 + (e&7).
// Each wave owns one 16-wide j tile x 64 tokens (4 accumulators).
// All operand loads are hoisted ahead of the WMMAs so the scheduler can
// clause them and overlap the fetch latency across the 4 matrix ops.
// MODE 0: dst bf16 (B,H,N,DH)  — Q/K projection (per-lane contiguous store)
// MODE 1: dst bf16 (B,H,DH,N)  — V projection (LDS transpose, then b128 store)
// MODE 2: src bf16 ctx, dst f32 (T,D) — output projection (direct f32 store)
// ---------------------------------------------------------------------------
template <int MODE>
__global__ __launch_bounds__(256, 2)
void proj_kernel(const float* __restrict__ Xf, const __bf16* __restrict__ Xb,
                 const float* __restrict__ W, __bf16* __restrict__ dstb,
                 float* __restrict__ dstf) {
  __shared__ __bf16 lds[8][16][24];  // per-wave 16x16 transpose tile (padded)

  const int wave = threadIdx.x >> 5;
  const int lane = threadIdx.x & 31;
  const int half = lane >> 4;
  const int lm   = lane & 15;

  const int jbase = (blockIdx.y * 8 + wave) * 16;  // gridDim.y = 8 -> 64 j-tiles
  const int tok0  = blockIdx.x * 64;               // gridDim.x = 128

  const float* wrow = W + (size_t)(jbase + lm) * Dd;  // this lane's weight row

  v8f acc[4] = {v8f{}, v8f{}, v8f{}, v8f{}};

  for (int k = 0; k < Dd; k += 32) {
    if (k + 32 < Dd) __builtin_prefetch(wrow + k + 32, 0, 1);
    // gather all operands first ...
    v16bf a = cat16(cvt8(wrow + k + half * 8), cvt8(wrow + k + 16 + half * 8));
    v16bf b[4];
#pragma unroll
    for (int g = 0; g < 4; ++g) {
      const int tok = tok0 + g * 16 + lm;
      if constexpr (MODE == 2) {
        const __bf16* xr = Xb + (size_t)tok * Dd;
        b[g] = cat16(ld8bf(xr + k + half * 8), ld8bf(xr + k + 16 + half * 8));
      } else {
        const float* xr = Xf + (size_t)tok * Dd;
        b[g] = cat16(cvt8(xr + k + half * 8), cvt8(xr + k + 16 + half * 8));
      }
    }
    // ... then issue the matrix ops back-to-back
#pragma unroll
    for (int g = 0; g < 4; ++g) acc[g] = wmma_bf16(a, b[g], acc[g]);
  }

  // D-tile layout: lane holds token col = lm (per g), rows j = jbase+half*8+r.
#pragma unroll
  for (int g = 0; g < 4; ++g) {
    if constexpr (MODE == 0) {
      // 8 consecutive j == 8 consecutive dh within one head -> one 16B store
      const int tok = tok0 + g * 16 + lm;
      const int j0 = jbase + half * 8;
      const int h = j0 / DH, dh = j0 % DH;
      const int bi = tok / Nn, ni = tok % Nn;
      v8bf o;
#pragma unroll
      for (int r = 0; r < 8; ++r) o[r] = (__bf16)acc[g][r];
      *(v8bf*)(dstb + (((size_t)(bi * Hh + h) * Nn + ni) * DH + dh)) = o;
    } else if constexpr (MODE == 2) {
      // 8 consecutive f32 output features for this token
      const int tok = tok0 + g * 16 + lm;
      float* o = dstf + (size_t)tok * Dd + jbase + half * 8;
      *(v4f*)o       = v4f{acc[g][0], acc[g][1], acc[g][2], acc[g][3]};
      *(v4f*)(o + 4) = v4f{acc[g][4], acc[g][5], acc[g][6], acc[g][7]};
    } else {
      // MODE 1: transpose 16x16 tile through LDS, store rows over tokens
#pragma unroll
      for (int r = 0; r < 8; ++r)
        lds[wave][half * 8 + r][lm] = (__bf16)acc[g][r];
      // read row j_local = lm, token half -> contiguous 8 tokens
      const int j = jbase + lm;
      const int h = j / DH, dh = j % DH;
      const int tokr = tok0 + g * 16 + half * 8;
      const int bi = tokr / Nn, ni = tokr % Nn;
      v8bf o = *(const v8bf*)&lds[wave][lm][half * 8];
      *(v8bf*)(dstb + (((size_t)(bi * Hh + h) * DH + dh) * Nn + ni)) = o;
    }
  }
}

// ---------------------------------------------------------------------------
// Flash attention over one (b,h) 16-query tile per wave, 8 waves / block.
// All 8 waves of a block share the same (b,h): K/V 32-key chunks are staged
// into double-buffered LDS by the Tensor Data Mover (wave 0 issues the DMA,
// s_wait_tensorcnt + workgroup barrier publishes it), so each chunk is read
// from L2 exactly once per block instead of 8x, and compute of chunk i
// overlaps the DMA of chunk i+1.
// ST = K(16key x 32d) x QT(32d x 16q): D-layout (lane = query col, keys in
// VGPR rows) is exactly the A-layout needed for P x V -> no transpose, and
// softmax row stats are per-lane (+ one shfl_xor(16) to merge key halves).
// ---------------------------------------------------------------------------
__global__ __launch_bounds__(256, 2)
void attn_kernel(const __bf16* __restrict__ q, const __bf16* __restrict__ k,
                 const __bf16* __restrict__ vt, __bf16* __restrict__ ctx) {
  __shared__ __align__(16) __bf16 kbuf[2][32 * DH];  // 2 x 4KB (rows = keys)
  __shared__ __align__(16) __bf16 vbuf[2][DH * 32];  // 2 x 4KB (rows = dh)
  __shared__ __bf16 sctx[8][16][24];

  const int wave = threadIdx.x >> 5;
  const int lane = threadIdx.x & 31;
  const int half = lane >> 4;
  const int lm   = lane & 15;

  const int tile = blockIdx.x * 8 + wave;  // 8192 q-tiles
  const int qb   = (tile & 127) * 16;      // 128 tiles per (b,h)
  const int bh   = tile >> 7;              // b*H + h (uniform across block)

  // Q operand (B of the ST gemm), loaded once: lane = its query's row
  const __bf16* qrow = q + ((size_t)bh * Nn + qb + lm) * DH;
  const v16bf Bq0 = cat16(ld8bf(qrow + half * 8), ld8bf(qrow + 16 + half * 8));
  const v16bf Bq1 = cat16(ld8bf(qrow + 32 + half * 8), ld8bf(qrow + 48 + half * 8));

  const __bf16* kbase = k + (size_t)bh * Nn * DH;
  const __bf16* vbase = vt + (size_t)bh * DH * Nn;

  // Prologue: stage chunk 0 into buffer 0.
  if (wave == 0) {
    // K chunk: 32 rows x 64 bf16 are contiguous -> flat 4KB copy (512 x 8B)
    tdm_load_to_lds((unsigned int)(size_t)&kbuf[0][0], kbase, 512, 1, 512, 1, 512);
    // V chunk: 64 dh-rows x 64B, row stride N*2B = 512 units
    tdm_load_to_lds((unsigned int)(size_t)&vbuf[0][0], vbase, 512, DH, 8, DH, 512);
    __builtin_amdgcn_s_wait_tensorcnt(0);
  }
  __syncthreads();

  v8f acc[4] = {v8f{}, v8f{}, v8f{}, v8f{}};
  float m = -1e30f, l = 0.f;
  const float SC = 0.125f * 1.44269504088896f;  // (1/sqrt(DH)) * log2(e)

  for (int k0 = 0; k0 < Nn; k0 += 32) {
    const int buf = (k0 >> 5) & 1;
    const bool more = (k0 + 32) < Nn;
    if (wave == 0 && more) {  // async-stage next chunk into the other buffer
      tdm_load_to_lds((unsigned int)(size_t)&kbuf[buf ^ 1][0],
                      kbase + (size_t)(k0 + 32) * DH, 512, 1, 512, 1, 512);
      tdm_load_to_lds((unsigned int)(size_t)&vbuf[buf ^ 1][0],
                      vbase + (k0 + 32), 512, DH, 8, DH, 512);
    }

    // --- gather ALL LDS operands for this chunk up front (clause-able) ---
    const __bf16* kl0 = &kbuf[buf][0] + (size_t)lm * DH;
    const __bf16* kl1 = kl0 + 16 * DH;
    v16bf ka[4], bv[4];
    ka[0] = cat16(ld8bf(kl0 + half * 8), ld8bf(kl0 + 16 + half * 8));
    ka[1] = cat16(ld8bf(kl0 + 32 + half * 8), ld8bf(kl0 + 48 + half * 8));
    ka[2] = cat16(ld8bf(kl1 + half * 8), ld8bf(kl1 + 16 + half * 8));
    ka[3] = cat16(ld8bf(kl1 + 32 + half * 8), ld8bf(kl1 + 48 + half * 8));
#pragma unroll
    for (int t = 0; t < 4; ++t) {
      const __bf16* vl = &vbuf[buf][0] + (size_t)(t * 16 + lm) * 32;
      bv[t] = cat16(ld8bf(vl + half * 8), ld8bf(vl + 16 + half * 8));
    }

    // --- ST = K x QT, two 16-key tiles ---
    v8f s0 = {}, s1 = {};
    s0 = wmma_bf16(ka[0], Bq0, s0);
    s0 = wmma_bf16(ka[1], Bq1, s0);
    s1 = wmma_bf16(ka[2], Bq0, s1);
    s1 = wmma_bf16(ka[3], Bq1, s1);

    // --- online softmax (log2 domain), per-lane rows ---
    float cm = -1e30f;
#pragma unroll
    for (int i = 0; i < 8; ++i) {
      s0[i] *= SC; s1[i] *= SC;
      cm = fmaxf(cm, fmaxf(s0[i], s1[i]));
    }
    cm = fmaxf(cm, __shfl_xor(cm, 16));  // merge the two key half-sets
    const float mnew = fmaxf(m, cm);
    const float alpha = fast_exp2(m - mnew);
    float ps = 0.f;
    v16bf ap;  // P in A-layout: e<8 -> tile0 row e, e>=8 -> tile1 row e-8
#pragma unroll
    for (int i = 0; i < 8; ++i) {
      const float p0 = fast_exp2(s0[i] - mnew);
      const float p1 = fast_exp2(s1[i] - mnew);
      ps += p0 + p1;
      ap[i] = (__bf16)p0;
      ap[i + 8] = (__bf16)p1;
    }
    l = l * alpha + ps;
    m = mnew;

    // rescale ctx accumulators: row q = half*8+r needs alpha from lane q
#pragma unroll
    for (int r = 0; r < 8; ++r) {
      const float ar = __shfl(alpha, ((lane & 16) >> 1) + r);
#pragma unroll
      for (int t = 0; t < 4; ++t) acc[t][r] *= ar;
    }

    // --- ctx += P x V, four dh tiles ---
#pragma unroll
    for (int t = 0; t < 4; ++t) acc[t] = wmma_bf16(ap, bv[t], acc[t]);

    if (wave == 0 && more) __builtin_amdgcn_s_wait_tensorcnt(0);
    __syncthreads();  // publish next buffer / protect current from overwrite
  }

  // finalize: divide rows by total l, store (token, D) bf16 via LDS transpose
  const float lt = l + __shfl_xor(l, 16);
  const float linv = 1.0f / lt;
  float lr[8];
#pragma unroll
  for (int r = 0; r < 8; ++r) lr[r] = __shfl(linv, ((lane & 16) >> 1) + r);

  const int bi = bh / Hh, h = bh % Hh;
#pragma unroll
  for (int t = 0; t < 4; ++t) {
#pragma unroll
    for (int r = 0; r < 8; ++r)
      sctx[wave][half * 8 + r][lm] = (__bf16)(acc[t][r] * lr[r]);
    // lane reads row q=lm, 8 contiguous dh -> one 16B global store
    v8bf o = *(const v8bf*)&sctx[wave][lm][half * 8];
    const size_t tok = (size_t)bi * Nn + qb + lm;
    *(v8bf*)(ctx + tok * Dd + h * DH + t * 16 + half * 8) = o;
  }
}

// ---------------------------------------------------------------------------
extern "C" void kernel_launch(void* const* d_in, const int* in_sizes, int n_in,
                              void* d_out, int out_size, void* d_ws, size_t ws_size,
                              hipStream_t stream) {
  const float* Q  = (const float*)d_in[0];
  const float* K  = (const float*)d_in[1];
  const float* V  = (const float*)d_in[2];
  const float* Wq = (const float*)d_in[3];
  const float* Wk = (const float*)d_in[4];
  const float* Wv = (const float*)d_in[5];
  const float* Wo = (const float*)d_in[6];
  float* out = (float*)d_out;

  // Workspace: 4 x 16 MB bf16 tensors (= 64 MB)
  __bf16* qp = (__bf16*)d_ws;                 // (B,H,N,DH)
  __bf16* kp = qp + (size_t)Tt * Dd;          // (B,H,N,DH)
  __bf16* vp = kp + (size_t)Tt * Dd;          // (B,H,DH,N)
  __bf16* cx = vp + (size_t)Tt * Dd;          // (T, D)
  (void)in_sizes; (void)n_in; (void)out_size; (void)ws_size;

  dim3 blk(256);
  dim3 gproj(Tt / 64, Dd / (16 * 8));  // (128, 8)

  proj_kernel<0><<<gproj, blk, 0, stream>>>(Q, nullptr, Wq, qp, nullptr);
  proj_kernel<0><<<gproj, blk, 0, stream>>>(K, nullptr, Wk, kp, nullptr);
  proj_kernel<1><<<gproj, blk, 0, stream>>>(V, nullptr, Wv, vp, nullptr);

  attn_kernel<<<dim3((Bb * Hh * Nn / 16) / 8), blk, 0, stream>>>(qp, kp, vp, cx);

  proj_kernel<2><<<gproj, blk, 0, stream>>>(nullptr, cx, Wo, nullptr, out);
}
